// InfoNCELoss_57578331570624
// MI455X (gfx1250) — matrix operand order repaired and checked by reading, not compiled, not run on hardware
//
#include <hip/hip_runtime.h>
#include <hip/hip_bf16.h>

typedef _Float16 v16h __attribute__((ext_vector_type(16)));
typedef float    v8f  __attribute__((ext_vector_type(8)));

#define BATCH   4096
#define NUM_GO  30000
#define DIM     128
#define NTILES  1875          // 30000 / 16
#define NSPLIT  8
#define WAVES   8             // waves per workgroup
#define ROWS_PER_WG (WAVES * 16)

// ---------------------------------------------------------------------------
// helpers: load 8 consecutive f32 (32B aligned), scale, convert to f16 into
// dst[base..base+7]
// ---------------------------------------------------------------------------
__device__ __forceinline__ void load8_cvt(const float* __restrict__ p, float s,
                                          v16h& dst, int base) {
    const float4* q = (const float4*)p;
    float4 f0 = q[0];
    float4 f1 = q[1];
    dst[base + 0] = (_Float16)(f0.x * s);
    dst[base + 1] = (_Float16)(f0.y * s);
    dst[base + 2] = (_Float16)(f0.z * s);
    dst[base + 3] = (_Float16)(f0.w * s);
    dst[base + 4] = (_Float16)(f1.x * s);
    dst[base + 5] = (_Float16)(f1.y * s);
    dst[base + 6] = (_Float16)(f1.z * s);
    dst[base + 7] = (_Float16)(f1.w * s);
}

// ---------------------------------------------------------------------------
// zero the workspace partial arrays
// ---------------------------------------------------------------------------
__global__ void zero_ws_kernel(float* __restrict__ ws, int n) {
    int i = blockIdx.x * blockDim.x + threadIdx.x;
    if (i < n) ws[i] = 0.0f;
}

// ---------------------------------------------------------------------------
// main fused GEMM + exp-sum kernel
//   grid  = (32 row-groups, NSPLIT)
//   block = 256 threads = 8 waves; wave w owns rows r0 = bx*128 + w*16
// partials: allPart/posPart are [NSPLIT][BATCH]
//
// Roofline: mask (492 MB, int32) is streamed exactly once -> non-temporal
// loads so it does not evict F (15.4 MB, heavily reused) from the 192 MB L2.
// ---------------------------------------------------------------------------
__global__ __launch_bounds__(256)
void infonce_main_kernel(const float* __restrict__ P,
                         const float* __restrict__ F,
                         const int*   __restrict__ M,
                         const float* __restrict__ T,
                         float* __restrict__ allPart,
                         float* __restrict__ posPart) {
    const int lane = threadIdx.x & 31;
    const int wave = threadIdx.x >> 5;
    const int hlf  = lane >> 4;      // 0: lanes 0-15, 1: lanes 16-31
    const int l15  = lane & 15;
    const int r0   = blockIdx.x * ROWS_PER_WG + wave * 16;  // M-tile base row
    const int split = blockIdx.y;

    const float invT = __expf(-T[0]);   // logits scale = 1/exp(t)

    // -------- preload A fragments (16 rows x K=128 f16), scaled by invT ----
    // 16-bit A 16x32 layout: lanes 0-15 M=lane, lanes 16-31 M=lane-16.
    // v16h halfs 0..7  -> K = ks*32 + hlf*8 + 0..7
    // v16h halfs 8..15 -> K = ks*32 + 16 + hlf*8 + 0..7
    const int am = r0 + l15;
    v16h A[4];
#pragma unroll
    for (int ks = 0; ks < 4; ++ks) {
        const float* p1 = P + (size_t)am * DIM + ks * 32 + hlf * 8;
        load8_cvt(p1,      invT, A[ks], 0);
        load8_cvt(p1 + 16, invT, A[ks], 8);
    }

    float sAll[8], sPos[8];
#pragma unroll
    for (int v = 0; v < 8; ++v) { sAll[v] = 0.0f; sPos[v] = 0.0f; }

    // -------- stream N chunks ---------------------------------------------
    int it = 0;
    for (int c = split; c < NTILES; c += NSPLIT, ++it) {
        const int n0 = c * 16;

        // B fragment: 16-bit B 32x16 layout: lane n = l15,
        // v16h halfs 0..7 -> K = ks*32 + hlf*8 + t, halfs 8..15 -> K = ks*32 + 16 + hlf*8 + (t-8)
        const float* brow = F + (size_t)(n0 + l15) * DIM + hlf * 8;
        v16h B[4];
#pragma unroll
        for (int ks = 0; ks < 4; ++ks) {
            load8_cvt(brow + ks * 32,     1.0f, B[ks], 0);
            load8_cvt(brow + ks * 32 + 16, 1.0f, B[ks], 8);
        }

        v8f acc = {0.f, 0.f, 0.f, 0.f, 0.f, 0.f, 0.f, 0.f};
#pragma unroll
        for (int ks = 0; ks < 4; ++ks) {
            acc = __builtin_amdgcn_wmma_f32_16x16x32_f16(
                false, A[ks], false, B[ks], (short)0, acc, false, false);
        }

        // C/D layout: lane column N = l15; VGPR v -> row M = hlf*8 + v
        // mask is touched exactly once across the dispatch -> non-temporal
        const int* mp = M + (size_t)(r0 + hlf * 8) * NUM_GO + n0 + l15;
#pragma unroll
        for (int v = 0; v < 8; ++v) {
            int   mv = __builtin_nontemporal_load(mp + (size_t)v * NUM_GO);
            float e  = __expf(acc[v]);           // logits already scaled via A
            sAll[v] += e;
            sPos[v] += (mv > 0) ? e : 0.0f;
        }

        // keep the 8 waves of this WG loosely in lockstep so the shared
        // B-tile stays hot in WGP$/L2 (trip count uniform per block)
        if ((it & 3) == 3) __syncthreads();
    }

    // -------- reduce over the 16 lanes of each half (columns) -------------
#pragma unroll
    for (int v = 0; v < 8; ++v) {
#pragma unroll
        for (int off = 1; off < 16; off <<= 1) {
            sAll[v] += __shfl_xor(sAll[v], off, 32);
            sPos[v] += __shfl_xor(sPos[v], off, 32);
        }
    }

    if (l15 == 0) {
        const int rb = r0 + hlf * 8;
#pragma unroll
        for (int v = 0; v < 8; ++v) {
            __builtin_nontemporal_store(sAll[v],
                allPart + (size_t)split * BATCH + rb + v);
            __builtin_nontemporal_store(sPos[v],
                posPart + (size_t)split * BATCH + rb + v);
        }
    }
}

// ---------------------------------------------------------------------------
// finalize: combine NSPLIT partials per row, per-row loss, mean over batch
// single block, deterministic fixed-order reduction
// ---------------------------------------------------------------------------
__global__ __launch_bounds__(256)
void infonce_finalize_kernel(const float* __restrict__ allPart,
                             const float* __restrict__ posPart,
                             float* __restrict__ out) {
    __shared__ float red[256];
    float acc = 0.0f;
    for (int r = threadIdx.x; r < BATCH; r += 256) {
        float s = 0.0f, p = 0.0f;
#pragma unroll
        for (int j = 0; j < NSPLIT; ++j) {
            s += allPart[(size_t)j * BATCH + r];
            p += posPart[(size_t)j * BATCH + r];
        }
        if (p > 0.0f) {
            acc += -__logf(p / s + 1e-8f);
        }
    }
    red[threadIdx.x] = acc;
    __syncthreads();
#pragma unroll
    for (int off = 128; off > 0; off >>= 1) {
        if (threadIdx.x < off) red[threadIdx.x] += red[threadIdx.x + off];
        __syncthreads();
    }
    if (threadIdx.x == 0) out[0] = red[0] / (float)BATCH;
}

// ---------------------------------------------------------------------------
extern "C" void kernel_launch(void* const* d_in, const int* in_sizes, int n_in,
                              void* d_out, int out_size, void* d_ws, size_t ws_size,
                              hipStream_t stream) {
    const float* P = (const float*)d_in[0];   // [4096, 128]  f32
    const float* F = (const float*)d_in[1];   // [30000, 128] f32
    const int*   M = (const int*)d_in[2];     // [4096, 30000] i32
    const float* T = (const float*)d_in[3];   // scalar f32 (log-temperature)
    float* out = (float*)d_out;

    float* allPart = (float*)d_ws;                       // [NSPLIT][BATCH]
    float* posPart = allPart + (size_t)NSPLIT * BATCH;   // [NSPLIT][BATCH]

    const int nPartial = 2 * NSPLIT * BATCH;             // 65536 floats
    zero_ws_kernel<<<(nPartial + 255) / 256, 256, 0, stream>>>(allPart, nPartial);

    dim3 grid(BATCH / ROWS_PER_WG, NSPLIT);              // (32, 8)
    infonce_main_kernel<<<grid, 256, 0, stream>>>(P, F, M, T, allPart, posPart);

    infonce_finalize_kernel<<<1, 256, 0, stream>>>(allPart, posPart, out);
}
